// Decoder_11012296147275
// MI455X (gfx1250) — matrix operand order repaired and checked by reading, not compiled, not run on hardware
//
#include <hip/hip_runtime.h>

// ---------------------------------------------------------------------------
// MI455X (gfx1250) implementation.
//  - GRU: 4 persistent workgroups (one per layer), software-pipelined over
//    steps, weights held in VGPRs as WMMA A-fragments, matvec via
//    v_wmma_f32_16x16x32_f16 (vector in column 0 of B), inter-WG sync via
//    agent-scope acquire/release flags in L2.
//  - Heads: tiled f16 WMMA GEMMs, 64-row blocks staged in LDS, f32 accum.
// ---------------------------------------------------------------------------

typedef __attribute__((ext_vector_type(16))) _Float16 v16h;
typedef __attribute__((ext_vector_type(8)))  _Float16 v8h;
typedef __attribute__((ext_vector_type(8)))  float    v8f;

#define HDIM   128
#define GDIM   384   // 3*HDIM gate rows
#define NLAYER 4
#define NMAX   4096

// ---------------- workspace layout ----------------
static constexpr size_t WIH16_OFF   = 0;                                   // halves
static constexpr size_t WHH16_OFF   = WIH16_OFF + (size_t)NLAYER*GDIM*HDIM;
static constexpr size_t HW16_OFF    = WHH16_OFF + (size_t)NLAYER*GDIM*HDIM;
static constexpr size_t HW16_STRIDE = (size_t)HDIM*HDIM;                   // one weight slot
static constexpr size_t ENC16_OFF   = HW16_OFF + (size_t)3*5*HW16_STRIDE;
static constexpr size_t HALVES_TOTAL= ENC16_OFF + (size_t)NMAX*HDIM;
static constexpr size_t BUF32_BYTE_OFF = ((HALVES_TOTAL*2 + 255)/256)*256; // fp32 layer outputs
static constexpr size_t FLAGS_BYTE_OFF = BUF32_BYTE_OFF + (size_t)NLAYER*NMAX*HDIM*4;
static constexpr size_t NFLAGS         = (size_t)NLAYER*NMAX;

// ---------------- fragment loaders (16-bit WMMA layouts) ----------------
// A 16x32 f16: lane m (0-15): halves 0-15 = K[hi*16..hi*16+15]
__device__ inline v16h load_a_rowmajor(const _Float16* base, int ld, int m0, int k0) {
  const int l = threadIdx.x & 31;
  const int m = l & 15, hi = l >> 4;
  const _Float16* p = base + (size_t)(m0 + m) * ld + k0 + hi * 16;
  v8h lo  = *(const v8h*)p;
  v8h hi8 = *(const v8h*)(p + 8);
  v16h r;
#pragma unroll
  for (int i = 0; i < 8; ++i) { r[i] = lo[i]; r[i + 8] = hi8[i]; }
  return r;
}

// B 32x16 f16 holding W^T: B(k,n) = W[n0+n][k0+k]; lane n, K = hi*16 + j
__device__ inline v16h load_b_wt(const _Float16* W, int ld, int n0, int k0, int nvalid) {
  const int l = threadIdx.x & 31;
  const int n = l & 15, hi = l >> 4;
  v16h r;
  if (n0 + n < nvalid) {
    const _Float16* p = W + (size_t)(n0 + n) * ld + k0 + hi * 16;
    v8h a = *(const v8h*)p;
    v8h b = *(const v8h*)(p + 8);
#pragma unroll
    for (int i = 0; i < 8; ++i) { r[i] = a[i]; r[i + 8] = b[i]; }
  } else {
#pragma unroll
    for (int i = 0; i < 16; ++i) r[i] = (_Float16)0.f;
  }
  return r;
}

// B 32x16 with the vector in column 0 only (matvec)
__device__ inline v16h load_b_vec(const _Float16* v, int k0) {
  const int l = threadIdx.x & 31;
  const int n = l & 15, hi = l >> 4;
  v16h r;
#pragma unroll
  for (int i = 0; i < 16; ++i) r[i] = (_Float16)0.f;
  if (n == 0) {
    const _Float16* p = v + k0 + hi * 16;
    v8h a = *(const v8h*)p;
    v8h b = *(const v8h*)(p + 8);
#pragma unroll
    for (int i = 0; i < 8; ++i) { r[i] = a[i]; r[i + 8] = b[i]; }
  }
  return r;
}

// ---------------- weight conversion ----------------
__global__ void cvt_f32_f16(_Float16* __restrict__ dst, const float* __restrict__ src, int n) {
  int i = blockIdx.x * blockDim.x + threadIdx.x;
  if (i < n) dst[i] = (_Float16)src[i];
}

// ---------------- GRU: 4 persistent WGs, layer-pipelined ----------------
__global__ void __launch_bounds__(384) gru_kernel(
    const float* __restrict__ x0,        // [128] initial h of layer 0
    const int*   __restrict__ n_steps_p,
    const float* __restrict__ b_ih,      // [4][384]
    const float* __restrict__ b_hh,      // [4][384]
    _Float16*    __restrict__ ws16,      // f16 region (weights + enc)
    float*       __restrict__ buf,       // [4][NMAX][128] layer outputs
    int*         __restrict__ flags)     // [4][NMAX]
{
  const int l    = blockIdx.x;           // layer
  const int tid  = threadIdx.x;
  const int wave = tid >> 5;
  const int lane = tid & 31;
  int n = *n_steps_p; if (n > NMAX) n = NMAX;

  __shared__ float     g[2 * GDIM];      // [gi(384) | gh(384)]
  __shared__ _Float16  xin16[HDIM];
  __shared__ _Float16  h16[HDIM];
  __shared__ float     h32[HDIM];

  // --- preload this layer's weights as register-resident A fragments ---
  const _Float16* wih = ws16 + WIH16_OFF + (size_t)l * GDIM * HDIM;
  const _Float16* whh = ws16 + WHH16_OFF + (size_t)l * GDIM * HDIM;
  v16h A[4][4];                           // [tile][kchunk], 128 VGPRs/lane
#pragma unroll
  for (int j = 0; j < 4; ++j) {
    const int gidx = wave * 4 + j;        // 0..23 -> W_ih, 24..47 -> W_hh
    const _Float16* wsrc = (gidx < 24) ? wih : whh;
    const int row0 = (gidx % 24) * 16;
#pragma unroll
    for (int c = 0; c < 4; ++c) A[j][c] = load_a_rowmajor(wsrc, HDIM, row0, c * 32);
  }
  const int mat = (wave < 6) ? 0 : 1;     // which source vector this wave consumes

  if (tid < HDIM) {
    float h0 = (l == 0) ? x0[tid] : 0.f;  // hid0[0] = x
    h32[tid] = h0; h16[tid] = (_Float16)h0;
  }
  __syncthreads();

  const float* bi = b_ih + l * GDIM;
  const float* bh = b_hh + l * GDIM;

  for (int t = 0; t < n; ++t) {
    // -------- obtain x_in from the upstream layer (pipeline) --------
    const int src_l = (l == 0) ? 3 : (l - 1);
    const int src_t = (l == 0) ? (t - 1) : t;
    if (src_t >= 0) {
      if (tid == 0) {
        while (__hip_atomic_load(&flags[src_l * NMAX + src_t], __ATOMIC_ACQUIRE,
                                 __HIP_MEMORY_SCOPE_AGENT) == 0) {
          __builtin_amdgcn_s_sleep(1);
        }
      }
      __syncthreads();
      if (tid < HDIM)
        xin16[tid] = (_Float16)buf[((size_t)src_l * NMAX + src_t) * HDIM + tid];
    } else {
      if (tid < HDIM) xin16[tid] = (_Float16)0.f;   // inp0 = 0
    }
    __syncthreads();

    // -------- gi = W_ih@x, gh = W_hh@h via WMMA (vector in B column 0) --------
    const _Float16* src16 = (mat == 0) ? xin16 : h16;
    v8f acc[4];
#pragma unroll
    for (int j = 0; j < 4; ++j)
#pragma unroll
      for (int i = 0; i < 8; ++i) acc[j][i] = 0.f;

#pragma unroll
    for (int c = 0; c < 4; ++c) {
      v16h B = load_b_vec(src16, c * 32);
#pragma unroll
      for (int j = 0; j < 4; ++j)
        acc[j] = __builtin_amdgcn_wmma_f32_16x16x32_f16(
            false, A[j][c], false, B, (short)0, acc[j], false, false);
    }

    // -------- extract column 0 (lanes 0 and 16 hold it) --------
    if ((lane & 15) == 0) {
      const int hi = lane >> 4;
#pragma unroll
      for (int j = 0; j < 4; ++j) {
        const int gidx = wave * 4 + j;
        const int base = ((gidx < 24) ? 0 : GDIM) + (gidx % 24) * 16 + hi * 8;
#pragma unroll
        for (int i = 0; i < 8; ++i) g[base + i] = acc[j][i];
      }
    }
    __syncthreads();

    // -------- gates (fp32) --------
    if (tid < HDIM) {
      float ir = g[tid]            + bi[tid];
      float iz = g[HDIM + tid]     + bi[HDIM + tid];
      float in_= g[2 * HDIM + tid] + bi[2 * HDIM + tid];
      float hr = g[GDIM + tid]            + bh[tid];
      float hz = g[GDIM + HDIM + tid]     + bh[HDIM + tid];
      float hn = g[GDIM + 2 * HDIM + tid] + bh[2 * HDIM + tid];
      float r  = 1.f / (1.f + expf(-(ir + hr)));
      float z  = 1.f / (1.f + expf(-(iz + hz)));
      float nn = tanhf(in_ + r * hn);
      float h  = (1.f - z) * nn + z * h32[tid];
      h32[tid] = h;
      h16[tid] = (_Float16)h;
      buf[((size_t)l * NMAX + t) * HDIM + tid] = h;          // fp32 for next layer
      if (l == 3) ws16[ENC16_OFF + (size_t)t * HDIM + tid] = (_Float16)h;  // enc for heads
      __threadfence();                                        // device-visible before flag
    }
    __syncthreads();
    if (tid == 0)
      __hip_atomic_store(&flags[l * NMAX + t], 1, __ATOMIC_RELEASE,
                         __HIP_MEMORY_SCOPE_AGENT);
  }
}

// ---------------- heads: tiled f16 WMMA GEMMs ----------------
struct HeadsArgs {
  const float* b[3][5];   // b1..b4, out_b per head
  int outdim[3];          // 128, 33, 128
  int outmult[3];         // output offset = n * outmult
};

// C = leaky( A(64x128 LDS) @ W^T + bias [+ res] ), dst f16 in LDS
__device__ inline void gemm64(const _Float16* __restrict__ Asrc,
                              const _Float16* __restrict__ Wk,
                              const float* __restrict__ bias,
                              _Float16* __restrict__ dst,
                              const _Float16* __restrict__ res) {
  const int wave = threadIdx.x >> 5, lane = threadIdx.x & 31;
  const int rm = wave >> 1;               // row tile 0..3
  const int ctbase = (wave & 1) * 4;      // col tiles ctbase..ctbase+3
  const int nlo = lane & 15, hi = lane >> 4;

  v8f acc[4];
#pragma unroll
  for (int j = 0; j < 4; ++j) {
    const float bv = bias[(ctbase + j) * 16 + nlo];
#pragma unroll
    for (int i = 0; i < 8; ++i) acc[j][i] = bv;
  }
#pragma unroll
  for (int c = 0; c < 4; ++c) {
    v16h Af = load_a_rowmajor(Asrc, HDIM, rm * 16, c * 32);
#pragma unroll
    for (int j = 0; j < 4; ++j) {
      v16h Bf = load_b_wt(Wk, HDIM, (ctbase + j) * 16, c * 32, HDIM);
      acc[j] = __builtin_amdgcn_wmma_f32_16x16x32_f16(
          false, Af, false, Bf, (short)0, acc[j], false, false);
    }
  }
#pragma unroll
  for (int j = 0; j < 4; ++j) {
    const int n0 = (ctbase + j) * 16;
#pragma unroll
    for (int i = 0; i < 8; ++i) {
      const int m = rm * 16 + hi * 8 + i;
      float v = acc[j][i];
      if (res) v += (float)res[m * HDIM + n0 + nlo];
      v = (v > 0.f) ? v : 0.2f * v;
      dst[m * HDIM + n0 + nlo] = (_Float16)v;
    }
  }
}

__device__ inline void gemm_out(const _Float16* __restrict__ Asrc,
                                const _Float16* __restrict__ Wk,
                                const float* __restrict__ bias,
                                float* __restrict__ outp, int outdim,
                                int s0, int rows) {
  const int wave = threadIdx.x >> 5, lane = threadIdx.x & 31;
  const int rm = wave >> 1;
  const int ctbase = (wave & 1) * 4;
  const int nlo = lane & 15, hi = lane >> 4;

#pragma unroll
  for (int j = 0; j < 4; ++j) {
    const int n0 = (ctbase + j) * 16;
    if (n0 >= outdim) continue;
    v8f acc;
    const float bv = (n0 + nlo < outdim) ? bias[n0 + nlo] : 0.f;
#pragma unroll
    for (int i = 0; i < 8; ++i) acc[i] = bv;
#pragma unroll
    for (int c = 0; c < 4; ++c) {
      v16h Af = load_a_rowmajor(Asrc, HDIM, rm * 16, c * 32);
      v16h Bf = load_b_wt(Wk, HDIM, n0, c * 32, outdim);
      acc = __builtin_amdgcn_wmma_f32_16x16x32_f16(
          false, Af, false, Bf, (short)0, acc, false, false);
    }
    if (n0 + nlo < outdim) {
#pragma unroll
      for (int i = 0; i < 8; ++i) {
        const int m = rm * 16 + hi * 8 + i;
        if (m < rows) outp[(size_t)(s0 + m) * outdim + n0 + nlo] = acc[i];
      }
    }
  }
}

__global__ void __launch_bounds__(256) heads_kernel(
    const int* __restrict__ n_steps_p,
    const _Float16* __restrict__ ws16,
    float* __restrict__ out,
    HeadsArgs ha) {
  int n = *n_steps_p; if (n > NMAX) n = NMAX;
  const int nblk = NMAX / 64;                 // 64 blocks per head
  const int hd  = blockIdx.x / nblk;
  const int sb0 = blockIdx.x % nblk;
  const int tid = threadIdx.x;

  __shared__ __align__(16) _Float16 xs[64 * HDIM];
  __shared__ __align__(16) _Float16 ys[64 * HDIM];

  const _Float16* enc = ws16 + ENC16_OFF;
  const _Float16* W[5];
#pragma unroll
  for (int i = 0; i < 5; ++i)
    W[i] = ws16 + HW16_OFF + ((size_t)hd * 5 + i) * HW16_STRIDE;
  const int outdim = ha.outdim[hd];
  float* outp = out + (size_t)n * ha.outmult[hd];

  for (int sb = sb0; sb * 64 < n; sb += nblk) {
    const int s0 = sb * 64;
    int rows = n - s0; if (rows > 64) rows = 64;

    for (int idx = tid; idx < 64 * HDIM; idx += 256) {
      const int r = idx >> 7, c = idx & 127;
      xs[idx] = (r < rows) ? enc[(size_t)(s0 + r) * HDIM + c] : (_Float16)0.f;
    }
    __syncthreads();

    gemm64(xs, W[0], ha.b[hd][0], ys, nullptr);  __syncthreads();  // res1 lin1 + leaky
    gemm64(ys, W[1], ha.b[hd][1], xs, xs);       __syncthreads();  // res1 lin2 + x + leaky
    gemm64(xs, W[2], ha.b[hd][2], ys, nullptr);  __syncthreads();  // res2 lin1 + leaky
    gemm64(ys, W[3], ha.b[hd][3], xs, xs);       __syncthreads();  // res2 lin2 + x + leaky
    gemm_out(xs, W[4], ha.b[hd][4], outp, outdim, s0, rows);
    __syncthreads();
  }
}

// ---------------- host-side launch ----------------
extern "C" void kernel_launch(void* const* d_in, const int* in_sizes, int n_in,
                              void* d_out, int out_size, void* d_ws, size_t ws_size,
                              hipStream_t stream) {
  (void)in_sizes; (void)n_in; (void)out_size; (void)ws_size;

  const float* x0      = (const float*)d_in[0];
  const int*   n_steps = (const int*)d_in[1];
  const float* w_ih    = (const float*)d_in[2];
  const float* w_hh    = (const float*)d_in[3];
  const float* b_ih    = (const float*)d_in[4];
  const float* b_hh    = (const float*)d_in[5];

  _Float16* ws16 = (_Float16*)d_ws;
  float* buf  = (float*)((char*)d_ws + BUF32_BYTE_OFF);
  int*   flags= (int*)  ((char*)d_ws + FLAGS_BYTE_OFF);

  // flags must start at 0 every call (producer/consumer handshake)
  hipMemsetAsync(flags, 0, NFLAGS * sizeof(int), stream);

  // fp32 -> f16 weight conversion
  const int CB = 256;
  {
    int ng = NLAYER * GDIM * HDIM;
    cvt_f32_f16<<<(ng + CB - 1) / CB, CB, 0, stream>>>(ws16 + WIH16_OFF, w_ih, ng);
    cvt_f32_f16<<<(ng + CB - 1) / CB, CB, 0, stream>>>(ws16 + WHH16_OFF, w_hh, ng);
  }
  HeadsArgs ha;
  const int outdims[3] = {128, 33, 128};
  const int outmult[3] = {0, 128, 161};  // a@0, p@n*128, m@n*(128+33)
  for (int h = 0; h < 3; ++h) {
    const int base = 6 + h * 10;
    for (int k = 0; k < 4; ++k) {        // w1..w4
      cvt_f32_f16<<<(HDIM * HDIM + CB - 1) / CB, CB, 0, stream>>>(
          ws16 + HW16_OFF + ((size_t)h * 5 + k) * HW16_STRIDE,
          (const float*)d_in[base + 2 * k], HDIM * HDIM);
      ha.b[h][k] = (const float*)d_in[base + 2 * k + 1];
    }
    const int no = outdims[h] * HDIM;    // out_w
    cvt_f32_f16<<<(no + CB - 1) / CB, CB, 0, stream>>>(
        ws16 + HW16_OFF + ((size_t)h * 5 + 4) * HW16_STRIDE,
        (const float*)d_in[base + 8], no);
    ha.b[h][4] = (const float*)d_in[base + 9];
    ha.outdim[h] = outdims[h];
    ha.outmult[h] = outmult[h];
  }

  // GRU: one workgroup per layer, pipelined over steps
  gru_kernel<<<NLAYER, 384, 0, stream>>>(x0, n_steps, b_ih, b_hh, ws16, buf, flags);

  // Heads: 3 heads x 64 step-blocks
  heads_kernel<<<3 * (NMAX / 64), 256, 0, stream>>>(n_steps, ws16, (float*)d_out, ha);
}
